// MultiHeadAttention_16114717295185
// MI455X (gfx1250) — matrix operand order repaired and checked by reading, not compiled, not run on hardware
//
#include <hip/hip_runtime.h>

typedef unsigned short u16t;
typedef __bf16 v16bf __attribute__((ext_vector_type(16)));
typedef __bf16 bfx8  __attribute__((ext_vector_type(8)));
typedef __bf16 bfx4  __attribute__((ext_vector_type(4)));
typedef float  v8f   __attribute__((ext_vector_type(8)));

#define TLEN   2048
#define DMODEL 1024
#define NH     16
#define DEPTH  64
#define NB     2
#define SC_LD  2064   // 2048 + 16 f32 pad

// ---------- CDNA5 async global->LDS copy (ASYNCcnt-tracked, bypasses VGPRs) ----------
__device__ __forceinline__ void async_b128(unsigned lds_off, const void* g) {
  asm volatile("global_load_async_to_lds_b128 %0, %1, off"
               :: "v"(lds_off), "v"(g) : "memory");
}
#define WAIT_ASYNC(n) asm volatile("s_wait_asynccnt %0" :: "n"(n) : "memory")

// ---------- WMMA fragment helpers (ISA 7.12.2 layouts) ----------
// A 16x32 bf16: lane<16 row M=lane holds K={k0..k0+7, k0+16..k0+23}; lane>=16 shifted +8.
__device__ __forceinline__ v16bf load_a_frag(const __bf16* base, int ld, int lane, int k0) {
  const int l = lane & 15, hi = lane >> 4;
  const __bf16* r = base + l * ld + k0 + hi * 8;
  v16bf a;
  ((bfx8*)&a)[0] = *(const bfx8*)(r);
  ((bfx8*)&a)[1] = *(const bfx8*)(r + 16);
  return a;
}
// B 32x16 bf16: lane<16 column N=lane holds K={k0..k0+7, k0+16..k0+23}; lane>=16 shifted +8.
// LDS tile staged K-contiguous per column: row-major [N][K].
__device__ __forceinline__ v16bf load_b_frag(const __bf16* base, int ld, int lane, int k0) {
  const int n = lane & 15, hi = lane >> 4;
  const __bf16* r = base + n * ld + k0 + hi * 8;
  v16bf b;
  ((bfx8*)&b)[0] = *(const bfx8*)(r);
  ((bfx8*)&b)[1] = *(const bfx8*)(r + 16);
  return b;
}
__device__ __forceinline__ v8f wmma_bf16(v16bf a, v16bf b, v8f c) {
  return __builtin_amdgcn_wmma_f32_16x16x32_bf16(false, a, false, b, (short)0, c,
                                                 false, false);
}

// ---------- kernel 0: bulk f32 -> bf16 conversion (native RNE cvt) ----------
__global__ __launch_bounds__(256) void cvt_kernel(const float* __restrict__ in,
                                                  __bf16* __restrict__ out, int n) {
  const int i = (blockIdx.x * 256 + threadIdx.x) * 4;
  if (i < n) {
    float4 f = *(const float4*)(in + i);
    bfx4 o = {(__bf16)f.x, (__bf16)f.y, (__bf16)f.z, (__bf16)f.w};
    *(bfx4*)(out + i) = o;
  }
}

// ---------- kernel 1: Y[n,m] = (X[n,:].W[m,:] + bias[m])*scale -> bf16 [B,H,T,64] ----------
__global__ __launch_bounds__(256) void proj_kernel(
    const __bf16* __restrict__ X, const __bf16* __restrict__ W,
    const float* __restrict__ bias, __bf16* __restrict__ Yh, float scale) {
  __shared__ __bf16 sx[2][128 * 40];  // double-buffered X tile [128][32], stride 40
  __shared__ __bf16 sw[2][64 * 40];   // double-buffered W tile [64][32]

  const int tid = threadIdx.x, lane = tid & 31, wave = tid >> 5;
  const int wm = wave >> 1, wn = wave & 1;
  const int m0 = blockIdx.x * 128, n0 = blockIdx.y * 64;

  const int xrow = tid >> 1, xkof = (tid & 1) * 16;
  const int wrow = tid >> 2, wkof = (tid & 3) * 8;
  const __bf16* Xg = X + (size_t)(m0 + xrow) * DMODEL + xkof;
  const __bf16* Wg = W + (size_t)(n0 + wrow) * DMODEL + wkof;

  auto issue = [&](int k0, int buf) {  // 3 async b128 per thread
    unsigned xo = (unsigned)(size_t)&sx[buf][xrow * 40 + xkof];
    async_b128(xo, Xg + k0);
    async_b128(xo + 16, Xg + k0 + 8);
    unsigned wo_ = (unsigned)(size_t)&sw[buf][wrow * 40 + wkof];
    async_b128(wo_, Wg + k0);
  };

  issue(0, 0);
  v8f acc[2][2] = {};
  for (int k0 = 0; k0 < DMODEL; k0 += 32) {
    const int buf = (k0 >> 5) & 1;
    if (k0 + 32 < DMODEL) {  // pipeline: DMA next tile while computing this one
      issue(k0 + 32, buf ^ 1);
      WAIT_ASYNC(3);         // async loads complete in order -> current tile resident
    } else {
      WAIT_ASYNC(0);
    }
    __syncthreads();
    v16bf b0 = load_b_frag(&sw[buf][(wn * 32 + 0) * 40], 40, lane, 0);
    v16bf b1 = load_b_frag(&sw[buf][(wn * 32 + 16) * 40], 40, lane, 0);
#pragma unroll
    for (int mt = 0; mt < 2; ++mt) {
      v16bf a = load_a_frag(&sx[buf][(wm * 32 + mt * 16) * 40], 40, lane, 0);
      acc[mt][0] = wmma_bf16(a, b0, acc[mt][0]);
      acc[mt][1] = wmma_bf16(a, b1, acc[mt][1]);
    }
    __syncthreads();
  }

  const int nloc = lane & 15, hi = lane >> 4;
#pragma unroll
  for (int mt = 0; mt < 2; ++mt)
#pragma unroll
    for (int nt = 0; nt < 2; ++nt) {
      const int colg = n0 + wn * 32 + nt * 16 + nloc;
      const float bv = bias[colg];
      const int h = colg >> 6, dd = colg & 63;
#pragma unroll
      for (int r = 0; r < 8; ++r) {
        const int rowg = m0 + wm * 32 + mt * 16 + r + hi * 8;
        const int bb = rowg >> 11, t = rowg & (TLEN - 1);
        Yh[(((size_t)(bb * NH + h) * TLEN) + t) * DEPTH + dd] =
            (__bf16)((acc[mt][nt][r] + bv) * scale);
      }
    }
}

// ---------- kernel 2: fused scores + softmax + attn write + AV ----------
__global__ __launch_bounds__(256) void attn_kernel(
    const __bf16* __restrict__ Qh, const __bf16* __restrict__ Kh,
    const __bf16* __restrict__ Vh, float* __restrict__ attn,
    __bf16* __restrict__ Oh) {
  extern __shared__ char smem[];
  float* sc = (float*)smem;                        // 32 x SC_LD f32 (264 KB)
  __bf16* qs = (__bf16*)(smem + 32 * SC_LD * 4);   // 32 x 72
  __bf16* kv0 = qs + 32 * 72;                      // 128x72 (K buf0) / 64x136 (V^T)
  __bf16* kv1 = kv0 + 128 * 72;                    // 128x72 (K buf1)

  const int tid = threadIdx.x, lane = tid & 31, wave = tid >> 5;
  const int bh = blockIdx.x >> 6;
  const int q0 = (blockIdx.x & 63) * 32;
  const size_t hbase = (size_t)bh * TLEN * DEPTH;

  // async-stage Q tile 32x64 (scaled by 1/8 at projection)
  const int qrow = tid >> 3, qkof = (tid & 7) * 8;
  async_b128((unsigned)(size_t)(qs + qrow * 72 + qkof),
             Qh + hbase + (size_t)(q0 + qrow) * DEPTH + qkof);

  const int krow = tid >> 1, kkof = (tid & 1) * 32;
  auto issueK = [&](int kt, __bf16* dst) {  // 4 async b128 per thread
    const __bf16* g = Kh + hbase + (size_t)(kt + krow) * DEPTH + kkof;
    unsigned o = (unsigned)(size_t)(dst + krow * 72 + kkof);
#pragma unroll
    for (int c = 0; c < 4; ++c) async_b128(o + c * 16, g + c * 8);
  };
  issueK(0, kv0);

  // ---- scores: sc[32][2048] = Q K^T, DMA-pipelined over 128-key chunks ----
  for (int kt = 0; kt < TLEN; kt += 128) {
    __bf16* cur = ((kt >> 7) & 1) ? kv1 : kv0;
    if (kt + 128 < TLEN) {
      issueK(kt + 128, ((kt >> 7) & 1) ? kv0 : kv1);
      WAIT_ASYNC(4);  // newest 4 may still fly; Q + current chunk are resident
    } else {
      WAIT_ASYNC(0);
    }
    __syncthreads();
#pragma unroll
    for (int i = 0; i < 2; ++i) {  // 16 tiles (2M x 8N) over 8 waves
      const int tileid = wave * 2 + i;
      const int mt = tileid & 1, nt = tileid >> 1;
      v8f acc = {};
#pragma unroll
      for (int kk = 0; kk < DEPTH; kk += 32) {
        v16bf a = load_a_frag(qs + mt * 16 * 72, 72, lane, kk);
        v16bf b = load_b_frag(cur + nt * 16 * 72, 72, lane, kk);
        acc = wmma_bf16(a, b, acc);
      }
      const int nloc = lane & 15, hi = lane >> 4;
      float* srow = sc + (mt * 16 + hi * 8) * SC_LD + kt + nt * 16 + nloc;
#pragma unroll
      for (int r = 0; r < 8; ++r) srow[r * SC_LD] = acc[r];
    }
    __syncthreads();
  }

  // ---- softmax over each of 32 rows; write attn exactly once ----
#pragma unroll 1
  for (int rr = 0; rr < 4; ++rr) {
    const int row = wave * 4 + rr;
    float* srow = sc + row * SC_LD;
    float mx = -3.402823466e38f;
    for (int j = 0; j < TLEN / 32; ++j) mx = fmaxf(mx, srow[lane + j * 32]);
#pragma unroll
    for (int off = 16; off > 0; off >>= 1) mx = fmaxf(mx, __shfl_xor(mx, off, 32));
    float sum = 0.f;
    for (int j = 0; j < TLEN / 32; ++j) {
      float e = __expf(srow[lane + j * 32] - mx);
      srow[lane + j * 32] = e;
      sum += e;
    }
#pragma unroll
    for (int off = 16; off > 0; off >>= 1) sum += __shfl_xor(sum, off, 32);
    const float inv = 1.f / sum;
    float* arow = attn + ((size_t)bh * TLEN + q0 + row) * TLEN;
    for (int j = 0; j < TLEN / 32; ++j) {
      const float p = srow[lane + j * 32] * inv;
      srow[lane + j * 32] = p;
      arow[lane + j * 32] = p;  // coalesced single write of the 537MB tensor
    }
  }
  __syncthreads();

  // ---- out = attn @ V: one 16x16 tile per wave, K=2048 ----
  const int mt = wave & 1, nt = wave >> 1;
  v8f oacc = {};
  const int vkey = tid >> 1, vds = (tid & 1) * 32;
  __bf16* vt = kv0;  // V^T staged [dim][key], stride 136
  for (int kt = 0; kt < TLEN; kt += 128) {
    {
      __bf16 buf[32];
      const uint4* g = (const uint4*)(Vh + hbase + (size_t)(kt + vkey) * DEPTH + vds);
      ((uint4*)buf)[0] = g[0]; ((uint4*)buf)[1] = g[1];
      ((uint4*)buf)[2] = g[2]; ((uint4*)buf)[3] = g[3];
#pragma unroll
      for (int j = 0; j < 32; ++j) vt[(vds + j) * 136 + vkey] = buf[j];
    }
    if (kt + 128 < TLEN)
      __builtin_prefetch(Vh + hbase + (size_t)(kt + 128 + vkey) * DEPTH + vds, 0, 0);
    __syncthreads();
    const int arow0 = (mt * 16 + (lane & 15)) * SC_LD;
    const int hi = lane >> 4;
#pragma unroll
    for (int kk = 0; kk < 128; kk += 32) {
      union { v16bf v; __bf16 s[16]; } ua;  // probs f32 -> bf16 A fragment
      const float* p0 = sc + arow0 + kt + kk + hi * 8;
#pragma unroll
      for (int e = 0; e < 8; ++e) ua.s[e] = (__bf16)p0[e];
#pragma unroll
      for (int e = 0; e < 8; ++e) ua.s[8 + e] = (__bf16)p0[16 + e];
      v16bf b = load_b_frag(vt + nt * 16 * 136, 136, lane, kk);
      oacc = wmma_bf16(ua.v, b, oacc);
    }
    __syncthreads();
  }
  {
    const int nloc = lane & 15, hi = lane >> 4;
#pragma unroll
    for (int r = 0; r < 8; ++r) {
      const int m = q0 + mt * 16 + r + hi * 8;
      Oh[hbase + (size_t)m * DEPTH + nt * 16 + nloc] = (__bf16)oacc[r];
    }
  }
}

// ---------- kernel 3: out = merge_heads(Oh) @ wo^T + bo (f32 out) ----------
__global__ __launch_bounds__(256) void oproj_kernel(
    const __bf16* __restrict__ Ah, const __bf16* __restrict__ W,
    const float* __restrict__ bias, float* __restrict__ Y) {
  __shared__ __bf16 sx[2][128 * 40];
  __shared__ __bf16 sw[2][64 * 40];
  const int tid = threadIdx.x, lane = tid & 31, wave = tid >> 5;
  const int wm = wave >> 1, wn = wave & 1;
  const int m0 = blockIdx.x * 128, n0 = blockIdx.y * 64;

  const int xrow = tid >> 1, xkof = (tid & 1) * 16;
  const int rowg = m0 + xrow;
  const int bb = rowg >> 11, t = rowg & (TLEN - 1);
  const int wrow = tid >> 2, wkof = (tid & 3) * 8;
  const __bf16* Wg = W + (size_t)(n0 + wrow) * DMODEL + wkof;

  auto issue = [&](int k0, int buf) {
    const int kg = k0 + xkof;          // k = h*64+d; a 16-run never crosses a head
    const int h = kg >> 6, dd = kg & 63;
    const __bf16* g = Ah + (((size_t)(bb * NH + h) * TLEN) + t) * DEPTH + dd;
    unsigned xo = (unsigned)(size_t)&sx[buf][xrow * 40 + xkof];
    async_b128(xo, g);
    async_b128(xo + 16, g + 8);
    unsigned wo_ = (unsigned)(size_t)&sw[buf][wrow * 40 + wkof];
    async_b128(wo_, Wg + k0);
  };

  issue(0, 0);
  v8f acc[2][2] = {};
  for (int k0 = 0; k0 < DMODEL; k0 += 32) {
    const int buf = (k0 >> 5) & 1;
    if (k0 + 32 < DMODEL) { issue(k0 + 32, buf ^ 1); WAIT_ASYNC(3); }
    else WAIT_ASYNC(0);
    __syncthreads();
    v16bf b0 = load_b_frag(&sw[buf][(wn * 32 + 0) * 40], 40, lane, 0);
    v16bf b1 = load_b_frag(&sw[buf][(wn * 32 + 16) * 40], 40, lane, 0);
#pragma unroll
    for (int mt = 0; mt < 2; ++mt) {
      v16bf a = load_a_frag(&sx[buf][(wm * 32 + mt * 16) * 40], 40, lane, 0);
      acc[mt][0] = wmma_bf16(a, b0, acc[mt][0]);
      acc[mt][1] = wmma_bf16(a, b1, acc[mt][1]);
    }
    __syncthreads();
  }

  const int nloc = lane & 15, hi = lane >> 4;
#pragma unroll
  for (int mt = 0; mt < 2; ++mt)
#pragma unroll
    for (int nt = 0; nt < 2; ++nt) {
      const int colg = n0 + wn * 32 + nt * 16 + nloc;
      const float bvv = bias[colg];
#pragma unroll
      for (int r = 0; r < 8; ++r) {
        const int rg = m0 + wm * 32 + mt * 16 + r + hi * 8;
        Y[(size_t)rg * DMODEL + colg] = acc[mt][nt][r] + bvv;
      }
    }
}

// ---------- host ----------
extern "C" void kernel_launch(void* const* d_in, const int* in_sizes, int n_in,
                              void* d_out, int out_size, void* d_ws, size_t ws_size,
                              hipStream_t stream) {
  const float* q  = (const float*)d_in[0];
  const float* k  = (const float*)d_in[1];
  const float* v  = (const float*)d_in[2];
  const float* wq = (const float*)d_in[3];
  const float* bq = (const float*)d_in[4];
  const float* wk = (const float*)d_in[5];
  const float* bk = (const float*)d_in[6];
  const float* wv = (const float*)d_in[7];
  const float* bv = (const float*)d_in[8];
  const float* wo = (const float*)d_in[9];
  const float* bo = (const float*)d_in[10];

  float* out  = (float*)d_out;
  float* attn = out + (size_t)NB * TLEN * DMODEL;

  const size_t nIn = (size_t)NB * TLEN * DMODEL;  // 4,194,304
  const size_t nW  = (size_t)DMODEL * DMODEL;     // 1,048,576
  __bf16* xq  = (__bf16*)d_ws;
  __bf16* xk  = xq + nIn;
  __bf16* xv  = xk + nIn;
  __bf16* wqb = xv + nIn;
  __bf16* wkb = wqb + nW;
  __bf16* wvb = wkb + nW;
  __bf16* wob = wvb + nW;
  __bf16* qh  = wob + nW;
  __bf16* kh  = qh + nIn;
  __bf16* vh  = kh + nIn;
  __bf16* oh  = vh + nIn;

  dim3 blk(256);
  cvt_kernel<<<(int)(nIn / 1024), blk, 0, stream>>>(q, xq, (int)nIn);
  cvt_kernel<<<(int)(nIn / 1024), blk, 0, stream>>>(k, xk, (int)nIn);
  cvt_kernel<<<(int)(nIn / 1024), blk, 0, stream>>>(v, xv, (int)nIn);
  cvt_kernel<<<(int)(nW / 1024), blk, 0, stream>>>(wq, wqb, (int)nW);
  cvt_kernel<<<(int)(nW / 1024), blk, 0, stream>>>(wk, wkb, (int)nW);
  cvt_kernel<<<(int)(nW / 1024), blk, 0, stream>>>(wv, wvb, (int)nW);
  cvt_kernel<<<(int)(nW / 1024), blk, 0, stream>>>(wo, wob, (int)nW);

  dim3 gp(32, 16);
  proj_kernel<<<gp, blk, 0, stream>>>(xq, wqb, bq, qh, 0.125f);  // 1/sqrt(64) folded
  proj_kernel<<<gp, blk, 0, stream>>>(xk, wkb, bk, kh, 1.0f);
  proj_kernel<<<gp, blk, 0, stream>>>(xv, wvb, bv, vh, 1.0f);

  const size_t smem = (size_t)32 * SC_LD * 4 + 32 * 72 * 2 + 2 * 128 * 72 * 2;  // ~306 KB
  (void)hipFuncSetAttribute((const void*)attn_kernel,
                            hipFuncAttributeMaxDynamicSharedMemorySize, (int)smem);
  attn_kernel<<<NB * NH * (TLEN / 32), blk, smem, stream>>>(qh, kh, vh, attn, oh);

  oproj_kernel<<<gp, blk, 0, stream>>>(oh, wob, bo, out);
}
